// PoseModel_35579509080130
// MI455X (gfx1250) — matrix operand order repaired and checked by reading, not compiled, not run on hardware
//
#include <hip/hip_runtime.h>
#include <hip/hip_bf16.h>

typedef __attribute__((ext_vector_type(16))) _Float16 v16h;
typedef __attribute__((ext_vector_type(8)))  _Float16 v8h;
typedef __attribute__((ext_vector_type(8)))  float    v8f;

#define B_  8
#define T_  256
#define C_  4
#define F_  256
#define FILT 128
#define G_  512          // 4*FILT
#define KK_ 384          // 3*FILT
#define HP_ (F_ + 2)     // h rows per batch incl. zero guard rows f=-1,256
#define EPS 1e-3f

__device__ __forceinline__ float hsig(float x) {
    return fminf(fmaxf(x * 0.2f + 0.5f, 0.0f), 1.0f);
}

// ---------------------------------------------------------------------------
// prep: zero c-state and BOTH h ping-pong buffers (incl. guard rows),
// build WhT[g][kk] (f16) from Wh[kk][g] (f32)
// ---------------------------------------------------------------------------
__global__ void convlstm_prep(const float* __restrict__ Wh,
                              float* __restrict__ c_state,
                              _Float16* __restrict__ h0,
                              _Float16* __restrict__ h1,
                              _Float16* __restrict__ WhT) {
    int idx = blockIdx.x * blockDim.x + threadIdx.x;
    if (idx < B_ * HP_ * FILT) {           // 264192: full padded h buffers
        h0[idx] = (_Float16)0.0f;
        h1[idx] = (_Float16)0.0f;
    }
    if (idx < B_ * FILT * F_)              // 262144
        c_state[idx] = 0.0f;
    if (idx < KK_ * G_) {                  // 196608
        int g  = idx % G_;
        int kk = idx / G_;
        WhT[(size_t)g * KK_ + kk] = (_Float16)Wh[(size_t)kk * G_ + g];
    }
}

// ---------------------------------------------------------------------------
// one recurrence step: gates = xconv(x_t) + b + conv(h_{t-1}, Wh); update c,h
// hprev/hnext layout: [b][f(-1..256)][cin], cin innermost, guard rows zero ->
// A-fragments are contiguous unconditional 16B loads at compile-time offsets
// from per-row base pointers; K-loop is explicitly double-buffered so loads
// of chunk kc+1 overlap the 8 WMMAs of chunk kc.
// grid: (32 M-tiles, 4 filter-groups), 128 threads (4 waves; wave == gate idx)
// ---------------------------------------------------------------------------
__global__ void __launch_bounds__(128, 1)
convlstm_step(const float* __restrict__ x,
              const float* __restrict__ Wx,
              const float* __restrict__ bias,
              const _Float16* __restrict__ WhT,
              const _Float16* __restrict__ hprev,
              _Float16* __restrict__ hnext,
              float* __restrict__ c_state,
              float* __restrict__ y,
              int t) {
    __shared__ float lds_g[64 * 32 * 4];   // [m][filt][gate] exchange buffer
    __shared__ float lds_x[C_ * 66];       // x halo slice [c][f-fBase+1]
    __shared__ float lds_w[12 * 128];      // Wx slice [(k*4+c)][gg*32+filt]

    const int tid   = threadIdx.x;
    const int wave  = tid >> 5;            // gate group: 0=i 1=f 2=c 3=o
    const int lane  = tid & 31;
    const int nl    = lane & 15;
    const int kHalf = lane >> 4;

    const int mTile = blockIdx.x;                 // 0..31
    const int nGrp  = blockIdx.y;                 // 0..3
    const int b     = mTile >> 2;                 // 4 tiles of 64 per batch row
    const int fBase = (mTile & 3) * 64;
    const int filtBase = nGrp * 32;
    const int gBase = wave * FILT + filtBase;     // global gate-channel base

    // stage x halo slice + Wx slice for the epilogue input-conv
    for (int e = tid; e < C_ * 66; e += 128) {
        int c  = e / 66;
        int fl = e % 66;                          // f = fBase + fl - 1
        int f  = fBase + fl - 1;
        float v = 0.0f;
        if (f >= 0 && f < F_)
            v = x[(((size_t)b * T_ + t) * C_ + c) * F_ + f];
        lds_x[c * 66 + fl] = v;
    }
    for (int e = tid; e < 12 * 128; e += 128) {
        int filt = e & 31;
        int rest = e >> 5;                        // (k*4+c)*4 + gg
        int gg   = rest & 3;
        int kc4  = rest >> 2;                     // k*4+c
        lds_w[kc4 * 128 + gg * 32 + filt] =
            Wx[(size_t)kc4 * G_ + gg * FILT + filtBase + filt];
    }

    // hoisted base pointers: all per-chunk deltas are compile-time constants
    const _Float16* pA[4];
    #pragma unroll
    for (int mt = 0; mt < 4; ++mt)
        pA[mt] = hprev + ((size_t)b * HP_ + fBase + mt * 16 + nl + 1) * FILT
                       + kHalf * 16;                   // dk=0 row position
    const _Float16* pB[2];
    #pragma unroll
    for (int nt = 0; nt < 2; ++nt)
        pB[nt] = WhT + (size_t)(gBase + nt * 16 + nl) * KK_ + kHalf * 16;

    union AFrag { v16h v; v8h h[2]; };
    AFrag a[2][4];
    v16h  bf[2][2];
    v8f   acc[4][2] = {};   // [mTileLocal][nTileLocal], 16x16 f32 each

    auto loadFrags = [&](int buf, int kc) {
        const int cinOff = ((kc >> 2) - 1) * FILT + (kc & 3) * 32; // element off
        #pragma unroll
        for (int mt = 0; mt < 4; ++mt) {
            a[buf][mt].h[0] = *(const v8h*)(pA[mt] + cinOff);
            a[buf][mt].h[1] = *(const v8h*)(pA[mt] + cinOff + 8);
        }
        #pragma unroll
        for (int nt = 0; nt < 2; ++nt)
            bf[buf][nt] = *(const v16h*)(pB[nt] + kc * 32);
    };

    loadFrags(0, 0);
    #pragma unroll
    for (int kc = 0; kc < 12; ++kc) {             // 12 chunks of 32 over KK=384
        const int cur = kc & 1;
        if (kc < 11)
            loadFrags(cur ^ 1, kc + 1);           // prefetch next chunk
        #pragma unroll
        for (int mt = 0; mt < 4; ++mt)
            #pragma unroll
            for (int nt = 0; nt < 2; ++nt)
                acc[mt][nt] = __builtin_amdgcn_wmma_f32_16x16x32_f16(
                    false, a[cur][mt].v, false, bf[cur][nt], (short)0,
                    acc[mt][nt], false, false);
    }

    // scatter D tiles into gate buffer: lds_g[m][filt][gate]
    #pragma unroll
    for (int mt = 0; mt < 4; ++mt)
        #pragma unroll
        for (int nt = 0; nt < 2; ++nt)
            #pragma unroll
            for (int r = 0; r < 8; ++r) {
                int M = mt * 16 + r + kHalf * 8;
                int N = nt * 16 + nl;
                lds_g[(M * 32 + N) * 4 + wave] = acc[mt][nt][r];
            }
    __syncthreads();

    // epilogue: add input conv + bias, activations, state update
    for (int e = tid; e < 64 * 32; e += 128) {
        int m    = e & 63;                        // consecutive f -> coalesced
        int filt = e >> 6;
        int f    = fBase + m;
        int fg   = filtBase + filt;               // global filter 0..127

        float g4[4];
        #pragma unroll
        for (int gg = 0; gg < 4; ++gg) {
            float s = bias[gg * FILT + fg] + lds_g[(m * 32 + filt) * 4 + gg];
            #pragma unroll
            for (int k = 0; k < 3; ++k)
                #pragma unroll
                for (int c = 0; c < C_; ++c)
                    s += lds_x[c * 66 + m + k] *
                         lds_w[(k * C_ + c) * 128 + gg * 32 + filt];
            g4[gg] = s;
        }

        float ig  = hsig(g4[0]);
        float fgt = hsig(g4[1]);
        float cc  = tanhf(g4[2]);
        float og  = hsig(g4[3]);

        size_t cidx = ((size_t)b * FILT + fg) * F_ + f;
        float cn = fgt * c_state[cidx] + ig * cc;
        c_state[cidx] = cn;
        float h = og * tanhf(cn);
        hnext[((size_t)b * HP_ + f + 1) * FILT + fg] = (_Float16)h;
        y[(((size_t)b * T_ + t) * FILT + fg) * F_ + f] = h;
    }
}

// ---------------------------------------------------------------------------
// LayerNorm over last axis (32768), in place, eps = 1e-3
// ---------------------------------------------------------------------------
__global__ void __launch_bounds__(256)
convlstm_ln(float* __restrict__ y,
            const float* __restrict__ gamma,
            const float* __restrict__ beta) {
    const int D = FILT * F_;                      // 32768
    __shared__ float rs[8], rq[8];
    size_t base = (size_t)blockIdx.x * D;
    int tid = threadIdx.x, wv = tid >> 5, lane = tid & 31;

    float s = 0.0f, q = 0.0f;
    for (int i = tid; i < D; i += 256) {
        float v = y[base + i];
        s += v;
        q += v * v;
    }
    #pragma unroll
    for (int o = 16; o > 0; o >>= 1) {
        s += __shfl_xor(s, o, 32);
        q += __shfl_xor(q, o, 32);
    }
    if (lane == 0) { rs[wv] = s; rq[wv] = q; }
    __syncthreads();
    if (tid == 0) {
        float S = 0.0f, Q = 0.0f;
        for (int i = 0; i < 8; ++i) { S += rs[i]; Q += rq[i]; }
        rs[0] = S; rq[0] = Q;
    }
    __syncthreads();
    float mu  = rs[0] * (1.0f / D);
    float var = rq[0] * (1.0f / D) - mu * mu;
    float inv = rsqrtf(var + EPS);
    for (int i = tid; i < D; i += 256) {
        float v = y[base + i];
        y[base + i] = (v - mu) * inv * gamma[i] + beta[i];
    }
}

// ---------------------------------------------------------------------------
extern "C" void kernel_launch(void* const* d_in, const int* in_sizes, int n_in,
                              void* d_out, int out_size, void* d_ws, size_t ws_size,
                              hipStream_t stream) {
    const float* x     = (const float*)d_in[0];   // (8,256,4,256)
    const float* Wx    = (const float*)d_in[1];   // (3,4,512)
    const float* Wh    = (const float*)d_in[2];   // (3,128,512)
    const float* bias  = (const float*)d_in[3];   // (512,)
    const float* gamma = (const float*)d_in[4];   // (32768,)
    const float* beta  = (const float*)d_in[5];   // (32768,)
    float* y = (float*)d_out;                     // (8,256,32768)

    const size_t NH = (size_t)B_ * HP_ * FILT;    // 264192 (padded h buffer)
    char* ws = (char*)d_ws;
    float*    c_state = (float*)ws;                                   // 1 MB
    _Float16* hbuf0   = (_Float16*)(ws + (1u << 20));
    _Float16* hbuf1   = hbuf0 + NH;
    _Float16* WhT     = hbuf1 + NH;

    convlstm_prep<<<(int)((NH + 255) / 256), 256, 0, stream>>>(
        Wh, c_state, hbuf0, hbuf1, WhT);

    for (int t = 0; t < T_; ++t) {
        const _Float16* hprev = (t & 1) ? hbuf1 : hbuf0;
        _Float16*       hnext = (t & 1) ? hbuf0 : hbuf1;
        convlstm_step<<<dim3(32, 4), 128, 0, stream>>>(
            x, Wx, bias, WhT, hprev, hnext, c_state, y, t);
    }

    convlstm_ln<<<B_ * T_, 256, 0, stream>>>(y, gamma, beta);
}
